// NodeUpdateV1_33827162423512
// MI455X (gfx1250) — matrix operand order repaired and checked by reading, not compiled, not run on hardware
//
#include <hip/hip_runtime.h>

// ---------------------------------------------------------------------------
// GNN message-passing layer for MI455X (gfx1250), wave32 + WMMA.
//   msg  = relu([x[src] || edge_emb] @ W_msg + b_msg)        (edge GEMM, K=192)
//   mean = scatter_mean(msg, dst)                            (f32 atomics, L2)
//   h    = relu([x || mean] @ W_upd + b_upd)                 (node GEMM, K=256)
// GEMMs run as f32-via-bf16x2 split on v_wmma_f32_16x16x32_bf16.
// ---------------------------------------------------------------------------

typedef __bf16 v16bf __attribute__((ext_vector_type(16)));
typedef float  v8f   __attribute__((ext_vector_type(8)));

static __device__ __forceinline__ void split_bf16(float x, __bf16& hi, __bf16& lo) {
    __bf16 h = (__bf16)x;
    hi = h;
    lo = (__bf16)(x - (float)h);
}

// Stage W[K][128] (f32 row-major) into LDS as transposed bf16 hi/lo images
// laid out [n][k] (k contiguous, row stride K) so B fragments are contiguous.
template <int K>
static __device__ void stage_weights(const float* __restrict__ W,
                                     __bf16* __restrict__ sHi,
                                     __bf16* __restrict__ sLo) {
    for (int idx = threadIdx.x; idx < K * 128; idx += blockDim.x) {
        int k = idx >> 7;     // idx / 128
        int n = idx & 127;    // idx % 128
        __bf16 h, l;
        split_bf16(W[idx], h, l);
        sHi[n * K + k] = h;
        sLo[n * K + k] = l;
    }
    __syncthreads();
}

// One wave computes C[16][128] = A[16][K] * B[K][128] (+bias preloaded in acc).
// A-fragment layout (16-bit A 16x32): lane l holds row m=l&15, K run
// [K0 + (l>>4)*16, +16) (matches B).
// B-fragment layout (16-bit B 32x16): lane l holds col n=l&15, K run
// [K0 + (l>>4)*16, +16)  -> one contiguous 32B read from the [n][k] LDS image.
template <int K, class AFetch>
static __device__ __forceinline__ void wave_gemm(v8f acc[8],
                                                 const __bf16* __restrict__ sHi,
                                                 const __bf16* __restrict__ sLo,
                                                 int lane, AFetch&& fetchA) {
    const int m  = lane & 15;
    const int kh = lane >> 4;
    #pragma unroll
    for (int ks = 0; ks < K / 32; ++ks) {
        const int K0 = ks * 32;
        float av[16];
        fetchA(m, K0 + kh * 16,     &av[0]);   // 8 contiguous f32
        fetchA(m, K0 + kh * 16 + 8, &av[8]);   // 8 contiguous f32
        v16bf ahi, alo;
        #pragma unroll
        for (int j = 0; j < 16; ++j) {
            __bf16 h, l;
            split_bf16(av[j], h, l);
            ahi[j] = h;
            alo[j] = l;
        }
        #pragma unroll
        for (int nt = 0; nt < 8; ++nt) {
            const int boff = (nt * 16 + m) * K + K0 + kh * 16;
            v16bf bhi = *(const v16bf*)(sHi + boff);
            v16bf blo = *(const v16bf*)(sLo + boff);
            // f32 ~ hi*hi + lo*hi + hi*lo (lo*lo negligible), f32 accumulate
            acc[nt] = __builtin_amdgcn_wmma_f32_16x16x32_bf16(
                false, ahi, false, bhi, (short)0, acc[nt], false, false);
            acc[nt] = __builtin_amdgcn_wmma_f32_16x16x32_bf16(
                false, alo, false, bhi, (short)0, acc[nt], false, false);
            acc[nt] = __builtin_amdgcn_wmma_f32_16x16x32_bf16(
                false, ahi, false, blo, (short)0, acc[nt], false, false);
        }
    }
}

// ---------------------------------------------------------------------------
// Kernel 1: edge message GEMM + ReLU + scatter-add (+ per-dst counts).
// One wave per 16-edge tile; 8 waves per block; W_msg staged in LDS.
// ---------------------------------------------------------------------------
__global__ void __launch_bounds__(256)
edge_msg_scatter(const float* __restrict__ node_emb,
                 const long long* __restrict__ src,
                 const long long* __restrict__ dst,
                 const float* __restrict__ edge_emb,
                 const float* __restrict__ W_msg,
                 const float* __restrict__ b_msg,
                 float* __restrict__ summed,
                 float* __restrict__ counts,
                 long long E, long long nTiles) {
    extern __shared__ char smem[];
    __bf16* sHi = (__bf16*)smem;
    __bf16* sLo = sHi + 192 * 128;
    stage_weights<192>(W_msg, sHi, sLo);

    const int lane = threadIdx.x & 31;
    const long long tile = (long long)blockIdx.x * 8 + (threadIdx.x >> 5);
    if (tile >= nTiles) return;               // uniform per wave -> EXEC stays full
    const long long base = tile * 16;

    v8f acc[8];
    #pragma unroll
    for (int nt = 0; nt < 8; ++nt) {
        float b = b_msg[nt * 16 + (lane & 15)];
        v8f t = {b, b, b, b, b, b, b, b};
        acc[nt] = t;
    }

    auto fetchA = [&](int m, int k, float* v) {
        long long e = base + m;
        if (e >= E) e = E - 1;                // clamp (keeps EXEC uniform)
        const float* p = (k < 128)
            ? node_emb + (size_t)src[e] * 128 + k        // gathered source row
            : edge_emb + (size_t)e * 64 + (k - 128);     // edge features
        float4 f0 = *(const float4*)p;
        float4 f1 = *(const float4*)(p + 4);
        v[0] = f0.x; v[1] = f0.y; v[2] = f0.z; v[3] = f0.w;
        v[4] = f1.x; v[5] = f1.y; v[6] = f1.z; v[7] = f1.w;
    };
    wave_gemm<192>(acc, sHi, sLo, lane, fetchA);

    // per-dst edge counts (one lane per edge row)
    if (lane < 16) {
        long long e = base + lane;
        if (e < E) atomicAdd(&counts[(size_t)dst[e]], 1.0f);
    }
    // ReLU + scatter-add: C/D layout row m = r + 8*(lane>>4), col = nt*16+(lane&15)
    #pragma unroll
    for (int r = 0; r < 8; ++r) {
        int m = r + ((lane >> 4) << 3);
        long long e = base + m;
        if (e < E) {
            float* row = summed + (size_t)dst[e] * 128;
            #pragma unroll
            for (int nt = 0; nt < 8; ++nt)
                atomicAdd(&row[nt * 16 + (lane & 15)], fmaxf(acc[nt][r], 0.0f));
        }
    }
}

// ---------------------------------------------------------------------------
// Kernel 2: node update GEMM + ReLU. mean = summed * 1/max(count,1) folded
// into the A-fetch. One wave per 16-node tile.
// ---------------------------------------------------------------------------
__global__ void __launch_bounds__(256)
node_update(const float* __restrict__ node_emb,
            const float* __restrict__ summed,
            const float* __restrict__ counts,
            const float* __restrict__ W_upd,
            const float* __restrict__ b_upd,
            float* __restrict__ out,
            int N, int nTiles) {
    extern __shared__ char smem[];
    __bf16* sHi = (__bf16*)smem;
    __bf16* sLo = sHi + 256 * 128;
    stage_weights<256>(W_upd, sHi, sLo);

    const int lane = threadIdx.x & 31;
    const int tile = blockIdx.x * 8 + (threadIdx.x >> 5);
    if (tile >= nTiles) return;
    const int base = tile * 16;

    v8f acc[8];
    #pragma unroll
    for (int nt = 0; nt < 8; ++nt) {
        float b = b_upd[nt * 16 + (lane & 15)];
        v8f t = {b, b, b, b, b, b, b, b};
        acc[nt] = t;
    }

    auto fetchA = [&](int m, int k, float* v) {
        int n0 = base + m;
        if (n0 >= N) n0 = N - 1;
        if (k < 128) {
            const float* p = node_emb + (size_t)n0 * 128 + k;
            float4 f0 = *(const float4*)p;
            float4 f1 = *(const float4*)(p + 4);
            v[0] = f0.x; v[1] = f0.y; v[2] = f0.z; v[3] = f0.w;
            v[4] = f1.x; v[5] = f1.y; v[6] = f1.z; v[7] = f1.w;
        } else {
            float inv = 1.0f / fmaxf(counts[n0], 1.0f);   // scatter-mean scale
            const float* p = summed + (size_t)n0 * 128 + (k - 128);
            float4 f0 = *(const float4*)p;
            float4 f1 = *(const float4*)(p + 4);
            v[0] = f0.x * inv; v[1] = f0.y * inv; v[2] = f0.z * inv; v[3] = f0.w * inv;
            v[4] = f1.x * inv; v[5] = f1.y * inv; v[6] = f1.z * inv; v[7] = f1.w * inv;
        }
    };
    wave_gemm<256>(acc, sHi, sLo, lane, fetchA);

    #pragma unroll
    for (int r = 0; r < 8; ++r) {
        int m = r + ((lane >> 4) << 3);
        int n0 = base + m;
        if (n0 < N) {
            float* o = out + (size_t)n0 * 128;
            #pragma unroll
            for (int nt = 0; nt < 8; ++nt)
                o[nt * 16 + (lane & 15)] = fmaxf(acc[nt][r], 0.0f);
        }
    }
}

// ---------------------------------------------------------------------------
extern "C" void kernel_launch(void* const* d_in, const int* in_sizes, int n_in,
                              void* d_out, int out_size, void* d_ws, size_t ws_size,
                              hipStream_t stream) {
    const float*     node_emb = (const float*)d_in[0];
    const long long* eidx     = (const long long*)d_in[1];   // int64 [2, E]
    const float*     edge_emb = (const float*)d_in[2];
    const float*     W_msg    = (const float*)d_in[3];
    const float*     b_msg    = (const float*)d_in[4];
    const float*     W_upd    = (const float*)d_in[5];
    const float*     b_upd    = (const float*)d_in[6];
    float*           out      = (float*)d_out;

    const int       N = in_sizes[0] / 128;
    const long long E = (long long)in_sizes[1] / 2;
    const long long* src = eidx;
    const long long* dst = eidx + E;

    // workspace: summed [N,128] f32 + counts [N] f32 (zeroed each launch)
    float* summed = (float*)d_ws;
    float* counts = summed + (size_t)N * 128;
    hipMemsetAsync(d_ws, 0, ((size_t)N * 128 + (size_t)N) * sizeof(float), stream);

    const long long tiles1 = (E + 15) / 16;
    const int blocks1 = (int)((tiles1 + 7) / 8);
    const size_t sh1 = 2u * 192u * 128u * 2u;   // hi+lo bf16 images of W_msg
    edge_msg_scatter<<<blocks1, 256, sh1, stream>>>(
        node_emb, src, dst, edge_emb, W_msg, b_msg, summed, counts, E, tiles1);

    const int tiles2 = (N + 15) / 16;
    const int blocks2 = (tiles2 + 7) / 8;
    const size_t sh2 = 2u * 256u * 128u * 2u;   // hi+lo bf16 images of W_upd
    node_update<<<blocks2, 256, sh2, stream>>>(
        node_emb, summed, counts, W_upd, b_upd, out, N, tiles2);
}
